// AttentionHead_43069932044583
// MI455X (gfx1250) — compile-verified, matrix operand order checked by reading, not run on hardware
//
#include <hip/hip_runtime.h>
#include <hip/hip_bf16.h>

#define B_N    4
#define S_LEN  2048
#define D_EMB  2048
#define HD     128
#define KC     32
#define QSCALE 0.08838834764831845f  // 1/sqrt(128)

typedef __attribute__((ext_vector_type(16))) __bf16 v16bf;
typedef __attribute__((ext_vector_type(8)))  __bf16 v8bf;
typedef __attribute__((ext_vector_type(8)))  float  v8f;
typedef __attribute__((ext_vector_type(4)))  int    v4i;

#define AS1 __attribute__((address_space(1)))
#define AS3 __attribute__((address_space(3)))

#if __has_builtin(__builtin_amdgcn_global_load_async_to_lds_b128)
#define HAVE_ASYNC_LDS 1
static __device__ inline void async_cp16(const void* g, void* l) {
  __builtin_amdgcn_global_load_async_to_lds_b128(
      (AS1 v4i*)g, (AS3 v4i*)l, 0, 0);
}
static __device__ inline void async_wait0() {
#if __has_builtin(__builtin_amdgcn_s_wait_asynccnt)
  __builtin_amdgcn_s_wait_asynccnt(0);
#else
  asm volatile("s_wait_asynccnt 0x0" ::: "memory");
#endif
}
#else
#define HAVE_ASYNC_LDS 0
#endif

static __device__ inline v16bf pack16(v8bf lo, v8bf hi) {
  v16bf r;
#pragma unroll
  for (int i = 0; i < 8; ++i) { r[i] = lo[i]; r[i + 8] = hi[i]; }
  return r;
}

// ---------------------------------------------------------------------------
// Kernel 1: fused QKV projection.  X is streamed ONCE (the HBM-dominant
// term); per K-chunk the three weight chunks are transposed to bf16 in LDS.
// Block = 256 thr = 8 waves; block tile = 64 rows x 128 cols x {Q,K,V}.
// Wave w: row group (w&3) -> rows [16*(w&3), +16), column half (w>>2).
// Per wave: 3 matrices x 4 column tiles = 12 v8f accumulators.
// Q is pre-scaled by 1/sqrt(HD); V stored transposed [B][HD][S].
// ---------------------------------------------------------------------------
__global__ __launch_bounds__(256) void qkv_fused_kernel(
    const float* __restrict__ X,
    const float* __restrict__ Wq, const float* __restrict__ Wk,
    const float* __restrict__ Wv,
    const float* __restrict__ bq, const float* __restrict__ bk,
    const float* __restrict__ bv,
    __bf16* __restrict__ Qo, __bf16* __restrict__ Ko,
    __bf16* __restrict__ Vt) {
  __shared__ __bf16 lds_wt[3][HD * KC];  // [m][n][k]  24 KB

  const int tid  = threadIdx.x;
  const int wave = tid >> 5;
  const int lane = tid & 31;
  const int hi   = lane >> 4;
  const int l16  = lane & 15;
  const int kb   = hi * 8;
  const int rg   = wave & 3;   // row group
  const int ch   = wave >> 2;  // column half

  const int rowBase = blockIdx.x * 64;
  const int arow    = rowBase + rg * 16 + l16;

  const float* Ws[3] = {Wq, Wk, Wv};

  v8f acc[3][4];
#pragma unroll
  for (int m = 0; m < 3; ++m)
#pragma unroll
    for (int tt = 0; tt < 4; ++tt) acc[m][tt] = (v8f)(0.0f);

  for (int kc = 0; kc < D_EMB; kc += KC) {
    __syncthreads();
    // Stage W*[kc:kc+32, 0:128] transposed into LDS as bf16.
#pragma unroll
    for (int m = 0; m < 3; ++m) {
#pragma unroll
      for (int it = 0; it < 4; ++it) {
        int idx = (tid + it * 256) * 4;
        int k = idx >> 7;
        int n = idx & 127;
        float4 wv = *(const float4*)(Ws[m] + (size_t)(kc + k) * HD + n);
        lds_wt[m][(n + 0) * KC + k] = (__bf16)wv.x;
        lds_wt[m][(n + 1) * KC + k] = (__bf16)wv.y;
        lds_wt[m][(n + 2) * KC + k] = (__bf16)wv.z;
        lds_wt[m][(n + 3) * KC + k] = (__bf16)wv.w;
      }
    }
    __syncthreads();

    // A fragment (shared by all 12 WMMAs this chunk).
    const float* xr = X + (size_t)arow * D_EMB + kc + kb;
    float4 x0 = *(const float4*)(xr);
    float4 x1 = *(const float4*)(xr + 4);
    float4 x2 = *(const float4*)(xr + 16);
    float4 x3 = *(const float4*)(xr + 20);
    v16bf a;
    a[0] = (__bf16)x0.x; a[1] = (__bf16)x0.y; a[2] = (__bf16)x0.z; a[3] = (__bf16)x0.w;
    a[4] = (__bf16)x1.x; a[5] = (__bf16)x1.y; a[6] = (__bf16)x1.z; a[7] = (__bf16)x1.w;
    a[8]  = (__bf16)x2.x; a[9]  = (__bf16)x2.y; a[10] = (__bf16)x2.z; a[11] = (__bf16)x2.w;
    a[12] = (__bf16)x3.x; a[13] = (__bf16)x3.y; a[14] = (__bf16)x3.z; a[15] = (__bf16)x3.w;

#pragma unroll
    for (int m = 0; m < 3; ++m) {
#pragma unroll
      for (int tt = 0; tt < 4; ++tt) {
        int n = ch * 64 + tt * 16 + l16;
        v16bf b = pack16(*(const v8bf*)(lds_wt[m] + n * KC + kb),
                         *(const v8bf*)(lds_wt[m] + n * KC + kb + 16));
        acc[m][tt] = __builtin_amdgcn_wmma_f32_16x16x32_bf16(
            false, a, false, b, (short)0, acc[m][tt], false, false);
      }
    }
  }

  // Epilogue: bias (+scale for Q), bf16 stores; V transposed.
  const float* Bs[3] = {bq, bk, bv};
#pragma unroll
  for (int m = 0; m < 3; ++m) {
#pragma unroll
    for (int tt = 0; tt < 4; ++tt) {
      int n = ch * 64 + tt * 16 + l16;
      float bias = Bs[m][n];
#pragma unroll
      for (int r = 0; r < 8; ++r) {
        float v = acc[m][tt][r] + bias;
        int mrow = rowBase + rg * 16 + r + hi * 8;
        if (m == 0) {
          Qo[(size_t)mrow * HD + n] = (__bf16)(v * QSCALE);
        } else if (m == 1) {
          Ko[(size_t)mrow * HD + n] = (__bf16)v;
        } else {
          int bb = mrow >> 11;
          int s  = mrow & (S_LEN - 1);
          Vt[((size_t)bb * HD + n) * S_LEN + s] = (__bf16)v;
        }
      }
    }
  }
}

// ---------------------------------------------------------------------------
// Kernel 2: causal flash attention.  Block = 256 thr = 8 waves; block owns a
// 128-query tile of one batch.  K and V^T tiles staged to LDS via the CDNA5
// async global->LDS path (ASYNCcnt) when available; online softmax in
// registers; P routed through LDS bf16 for C-layout -> A-layout conversion.
// Dynamic LDS: 32KB K + 32KB V + 32KB P = 96KB.
// ---------------------------------------------------------------------------
__global__ __launch_bounds__(256) void attn_kernel(
    const __bf16* __restrict__ Q,   // [B*S, HD] bf16, pre-scaled
    const __bf16* __restrict__ K,   // [B*S, HD] bf16
    const __bf16* __restrict__ Vt,  // [B, HD, S] bf16
    float* __restrict__ O) {        // [B, S, HD] fp32
  extern __shared__ char smem_raw[];
  __bf16* lds_k = (__bf16*)smem_raw;                // [key][d]   32 KB
  __bf16* lds_v = (__bf16*)(smem_raw + 32 * 1024);  // [d][key]   32 KB
  __bf16* lds_p = (__bf16*)(smem_raw + 64 * 1024);  // [q][key]   32 KB

  const int tid  = threadIdx.x;
  const int wave = tid >> 5;
  const int lane = tid & 31;
  const int hi   = lane >> 4;
  const int l16  = lane & 15;
  const int kb   = hi * 8;

  const int qt    = blockIdx.x;
  const int b     = blockIdx.y;
  const int qbase = qt * 128;

  // Q A-fragments, loaded once (d = 4 chunks of K=32).
  const int aqrow = qbase + wave * 16 + l16;
  v16bf qa[4];
#pragma unroll
  for (int kc = 0; kc < 4; ++kc) {
    const __bf16* qr = Q + ((size_t)b * S_LEN + aqrow) * HD + kc * 32 + kb;
    qa[kc] = pack16(*(const v8bf*)(qr), *(const v8bf*)(qr + 16));
  }

  v8f oacc[8];
#pragma unroll
  for (int t = 0; t < 8; ++t) oacc[t] = (v8f)(0.0f);
  float mrow[8], lrow[8];
#pragma unroll
  for (int r = 0; r < 8; ++r) { mrow[r] = -3.0e38f; lrow[r] = 0.0f; }

  // Per-thread V staging slice: half a V^T row (64 keys = 128 B).
  const int vrow  = tid >> 1;
  const int vkoff = (tid & 1) * 64;

  for (int kt = 0; kt <= qt; ++kt) {
    const int kbase = kt * 128;
    const __bf16* ksrc = K + ((size_t)b * S_LEN + kbase) * HD;
    const __bf16* vsrc =
        Vt + ((size_t)b * HD + vrow) * S_LEN + kbase + vkoff;

    __syncthreads();
#if HAVE_ASYNC_LDS
#pragma unroll
    for (int it = 0; it < 8; ++it) {
      int e = (tid + it * 256) * 8;
      async_cp16(ksrc + e, lds_k + e);
      async_cp16(vsrc + it * 8, lds_v + vrow * 128 + vkoff + it * 8);
    }
#else
#pragma unroll
    for (int it = 0; it < 8; ++it) {
      int e = (tid + it * 256) * 8;
      *(v8bf*)(lds_k + e) = *(const v8bf*)(ksrc + e);
      *(v8bf*)(lds_v + vrow * 128 + vkoff + it * 8) =
          *(const v8bf*)(vsrc + it * 8);
    }
#endif
    if (kt < qt) __builtin_prefetch(ksrc + (size_t)128 * HD + tid * 64, 0, 1);
#if HAVE_ASYNC_LDS
    async_wait0();
#endif
    __syncthreads();

    // Scores: S = Q . K^T.
    v8f sacc[8];
#pragma unroll
    for (int t = 0; t < 8; ++t) sacc[t] = (v8f)(0.0f);
#pragma unroll
    for (int kc = 0; kc < 4; ++kc) {
#pragma unroll
      for (int t = 0; t < 8; ++t) {
        int n = t * 16 + l16;
        v16bf bb = pack16(*(const v8bf*)(lds_k + n * HD + kc * 32 + kb),
                          *(const v8bf*)(lds_k + n * HD + kc * 32 + kb + 16));
        sacc[t] = __builtin_amdgcn_wmma_f32_16x16x32_bf16(
            false, qa[kc], false, bb, (short)0, sacc[t], false, false);
      }
    }

    // Causal mask (diagonal tile only).
    if (kt == qt) {
#pragma unroll
      for (int t = 0; t < 8; ++t) {
        int key = t * 16 + l16;
#pragma unroll
        for (int r = 0; r < 8; ++r) {
          int qrow = wave * 16 + r + hi * 8;
          if (key > qrow) sacc[t][r] = -1.0e30f;
        }
      }
    }

    // Online softmax (rows live in 16-lane halves of the wave).
#pragma unroll
    for (int r = 0; r < 8; ++r) {
      float mx = sacc[0][r];
#pragma unroll
      for (int t = 1; t < 8; ++t) mx = fmaxf(mx, sacc[t][r]);
#pragma unroll
      for (int off = 1; off < 16; off <<= 1) mx = fmaxf(mx, __shfl_xor(mx, off, 32));
      float mnew = fmaxf(mrow[r], mx);
      float corr = __expf(mrow[r] - mnew);
      mrow[r] = mnew;
      float sum = 0.0f;
#pragma unroll
      for (int t = 0; t < 8; ++t) {
        float p = __expf(sacc[t][r] - mnew);
        sacc[t][r] = p;
        sum += p;
      }
#pragma unroll
      for (int off = 1; off < 16; off <<= 1) sum += __shfl_xor(sum, off, 32);
      lrow[r] = lrow[r] * corr + sum;
#pragma unroll
      for (int t = 0; t < 8; ++t) oacc[t][r] *= corr;
    }

    // P -> LDS bf16 (C-layout regs to row-major for A-fragment loads).
    __syncthreads();
#pragma unroll
    for (int t = 0; t < 8; ++t) {
      int n = t * 16 + l16;
#pragma unroll
      for (int r = 0; r < 8; ++r) {
        int m = wave * 16 + r + hi * 8;
        lds_p[m * 128 + n] = (__bf16)sacc[t][r];
      }
    }
    __syncthreads();

    // O += P . V  (A = P rows, B column = d index from staged V^T).
#pragma unroll
    for (int kc2 = 0; kc2 < 4; ++kc2) {
      const __bf16* pr = lds_p + (wave * 16 + l16) * 128 + kc2 * 32 + kb;
      v16bf pa = pack16(*(const v8bf*)(pr), *(const v8bf*)(pr + 16));
#pragma unroll
      for (int t = 0; t < 8; ++t) {
        int nd = t * 16 + l16;
        const __bf16* vr = lds_v + nd * 128 + kc2 * 32 + kb;
        v16bf bb = pack16(*(const v8bf*)(vr), *(const v8bf*)(vr + 16));
        oacc[t] = __builtin_amdgcn_wmma_f32_16x16x32_bf16(
            false, pa, false, bb, (short)0, oacc[t], false, false);
      }
    }
  }

  // Epilogue: normalize and store fp32.
#pragma unroll
  for (int t = 0; t < 8; ++t) {
    int n = t * 16 + l16;
#pragma unroll
    for (int r = 0; r < 8; ++r) {
      int m = qbase + wave * 16 + r + hi * 8;
      O[((size_t)b * S_LEN + m) * HD + n] = oacc[t][r] / lrow[r];
    }
  }
}

extern "C" void kernel_launch(void* const* d_in, const int* in_sizes, int n_in,
                              void* d_out, int out_size, void* d_ws, size_t ws_size,
                              hipStream_t stream) {
  const float* X  = (const float*)d_in[0];
  const float* Wq = (const float*)d_in[1];
  const float* bq = (const float*)d_in[2];
  const float* Wk = (const float*)d_in[3];
  const float* bk = (const float*)d_in[4];
  const float* Wv = (const float*)d_in[5];
  const float* bv = (const float*)d_in[6];
  float* out = (float*)d_out;

  const size_t N = (size_t)B_N * S_LEN * HD;
  __bf16* Qbf = (__bf16*)d_ws;
  __bf16* Kbf = Qbf + N;
  __bf16* Vt  = Kbf + N;

  qkv_fused_kernel<<<(B_N * S_LEN) / 64, 256, 0, stream>>>(
      X, Wq, Wk, Wv, bq, bk, bv, Qbf, Kbf, Vt);
  attn_kernel<<<dim3(16, B_N), 256, 96 * 1024, stream>>>(Qbf, Kbf, Vt, out);
}
